// MLPDecoder_Causal_11639361372236
// MI455X (gfx1250) — matrix operand order repaired and checked by reading, not compiled, not run on hardware
//
#include <hip/hip_runtime.h>

namespace {

constexpr int An = 32;    // agents
constexpr int Tn = 64;    // timesteps
constexpr int Dn = 16;    // feature dim
constexpr int En = 992;   // edges = A*(A-1)

typedef __attribute__((ext_vector_type(16))) __bf16 v16bf;
typedef __attribute__((ext_vector_type(8)))  __bf16 v8bf;
typedef __attribute__((ext_vector_type(8)))  float  v8f;
typedef __attribute__((ext_vector_type(2)))  float  f32x2;
typedef __attribute__((ext_vector_type(2)))  __bf16 bf16x2;

union AF { v16bf v; v8bf h[2]; unsigned short u[16]; };
union CF { v8f  v; float f[8]; };

// native f32 -> bf16 (RNE); lets the backend pick hardware cvt
__device__ __forceinline__ unsigned short f2bf(float x) {
  union { __bf16 h; unsigned short u; } c;
  c.h = (__bf16)x;
  return c.u;
}

// ReLU + packed f32x2 -> bf16x2 conversion (v_cvt_pk_bf16_f32), two b16 stores
__device__ __forceinline__ void relu_pair_bf16(unsigned short* p0, unsigned short* p1,
                                               float x, float y) {
  f32x2 f;
  f.x = fmaxf(x, 0.0f);
  f.y = fmaxf(y, 0.0f);
  union { bf16x2 v; unsigned short us[2]; } c;
  c.v = __builtin_convertvector(f, bf16x2);
  *p0 = c.us[0];
  *p1 = c.us[1];
}

struct alignas(32) Smem {
  // Weight fragments in WMMA B-fragment order: [..][lane][element], 32B/lane row.
  // msg weights live during the edge phase; out weights overlay them afterwards.
  union {
    struct {
      unsigned short w1sw[2][4][32][16];      // (mk, n, lane, f)        8 KB
      unsigned short w2sw[2][2][4][32][16];   // (mk, kf, n, lane, f)   16 KB
    } msg;                                    // 24 KB
    struct {
      unsigned short o1sw[3][4][32][16];      // (kf, n, lane, f)       12 KB
      unsigned short o2sw[2][4][32][16];      // (kf, n, lane, f)        8 KB
      unsigned short o3sw[2][32][16];         // (kf, lane, f)           2 KB
    } outw;                                   // 22 KB
  } w;
  float          xs[32][16];               // node features f32       2 KB
  unsigned short xsb[32][16];              // node features bf16      1 KB
  float          b1[2][64];
  float          b2[2][64];
  float          ob1[64];
  float          ob2[64];
  float          ob3[16];
  float          agg[32][64];              // scatter-sum accum       8 KB
  unsigned short stage[8][1024];           // per-wave 16x64 buffers 16 KB
  float          c0[992];                  // edge coeff (k=0)       ~4 KB
};                                          // total ~57.8 KB

template <int STEP>
__global__ __launch_bounds__(256)
void nri_step_kernel(const float* __restrict__ inputs,
                     float* __restrict__ outp,
                     const float* __restrict__ rel_graph,
                     const float* __restrict__ gumbel,
                     const float* __restrict__ msg_W1,
                     const float* __restrict__ msg_b1,
                     const float* __restrict__ msg_W2,
                     const float* __restrict__ msg_b2,
                     const float* __restrict__ out_W1,
                     const float* __restrict__ out_b1,
                     const float* __restrict__ out_W2,
                     const float* __restrict__ out_b2,
                     const float* __restrict__ out_W3,
                     const float* __restrict__ out_b3) {
  __shared__ Smem s;
  const int tid = threadIdx.x;
  const int blk = blockIdx.x;
  const int b = blk >> 5;   // batch
  const int t = blk & 31;   // strided time index (0..31)

  // ---------- stage msg weights / activations / coefficients into LDS ----------
  for (int idx = tid; idx < 2 * 4 * 32 * 16; idx += 256) {
    int f = idx & 15, lanei = (idx >> 4) & 31, n = (idx >> 9) & 3, mk = idx >> 11;
    int kk  = (f >> 3) * 16 + (lanei >> 4) * 8 + (f & 7);
    int col = n * 16 + (lanei & 15);
    s.w.msg.w1sw[mk][n][lanei][f] = f2bf(msg_W1[(mk * 32 + kk) * 64 + col]);
  }
  for (int idx = tid; idx < 2 * 2 * 4 * 32 * 16; idx += 256) {
    int f = idx & 15, lanei = (idx >> 4) & 31, n = (idx >> 9) & 3;
    int kf = (idx >> 11) & 1, mk = idx >> 12;
    int kk  = kf * 32 + (f >> 3) * 16 + (lanei >> 4) * 8 + (f & 7);
    int col = n * 16 + (lanei & 15);
    s.w.msg.w2sw[mk][kf][n][lanei][f] = f2bf(msg_W2[(mk * 64 + kk) * 64 + col]);
  }
  for (int i = tid; i < 32 * 16; i += 256) {
    int a = i >> 4, d = i & 15;
    float v;
    if (STEP == 0) v = inputs[((b * An + a) * Tn + 2 * t) * Dn + d];       // x[:, ::2]
    else           v = outp[((b * An + a) * (Tn - 1) + 2 * t) * Dn + d];   // prev step (even tau)
    s.xs[a][d]  = v;
    s.xsb[a][d] = f2bf(v);
  }
  for (int i = tid; i < 128; i += 256) (&s.b1[0][0])[i] = msg_b1[i];
  for (int i = tid; i < 128; i += 256) (&s.b2[0][0])[i] = msg_b2[i];
  if (tid < 64) s.ob1[tid] = out_b1[tid];
  if (tid < 64) s.ob2[tid] = out_b2[tid];
  if (tid < 16) s.ob3[tid] = out_b3[tid];
  for (int i = tid; i < 32 * 64; i += 256) (&s.agg[0][0])[i] = 0.0f;
  for (int e = tid; e < En; e += 256) {
    // softmax over K=2 with tau=0.5  ->  sigmoid
    float l0 = (rel_graph[2 * e + 0] + gumbel[2 * e + 0]) * 2.0f;
    float l1 = (rel_graph[2 * e + 1] + gumbel[2 * e + 1]) * 2.0f;
    s.c0[e] = 1.0f / (1.0f + __expf(l1 - l0));
  }
  __syncthreads();

  const int wave = tid >> 5;
  const int lane = tid & 31;
  const int hl = lane >> 4;     // which K-half this lane carries (A/B frags)
  const int ln = lane & 15;     // row (A) / column (B,C,D) index

  // hoist biases into registers
  float bs1[2][4], bs2[2][4];
#pragma unroll
  for (int mk = 0; mk < 2; ++mk)
#pragma unroll
    for (int n = 0; n < 4; ++n) {
      bs1[mk][n] = s.b1[mk][n * 16 + ln];
      bs2[mk][n] = s.b2[mk][n * 16 + ln];
    }

  // ---------- edge message MLP: 62 M-tiles of 16 edges, round-robin over 8 waves ----------
  for (int mt = wave; mt < 62; mt += 8) {
    // A fragment = pre_msg tile (16 edges x 32 feats): [send(16) | recv(16)]
    const int e0 = mt * 16 + ln;
    const int i0 = e0 / 31;
    const int jr = e0 - i0 * 31;
    const int j0 = jr + (jr >= i0 ? 1 : 0);
    AF a;
    a.h[0] = *(const v8bf*)&s.xsb[j0][hl * 8];   // send features, K half hl
    a.h[1] = *(const v8bf*)&s.xsb[i0][hl * 8];   // recv features, K half hl

    // per-row edge coefficients (8 contiguous) + receiver segmentation
    const int ebase = mt * 16 + hl * 8;          // first edge of this half's rows
    const int iA = (mt * 16) / 31;               // receiver of tile's first edge
    const int eB = 31 * (iA + 1);                // boundary edge (start of iA+1)
    float c0v[8];
#pragma unroll
    for (int r = 0; r < 8; ++r) c0v[r] = s.c0[ebase + r];

    for (int mk = 0; mk < 2; ++mk) {   // edge-type mixture component
      // ---- layer 1: [16x32] @ [32x64] --- issue all 4 WMMAs, then post-process ----
      CF accA[4];
#pragma unroll
      for (int n = 0; n < 4; ++n) {
        AF w1f;
        w1f.v = *(const v16bf*)&s.w.msg.w1sw[mk][n][lane][0];   // contiguous 32B
        CF cin;
#pragma unroll
        for (int r = 0; r < 8; ++r) cin.f[r] = bs1[mk][n];
        accA[n].v = __builtin_amdgcn_wmma_f32_16x16x32_bf16(false, a.v, false, w1f.v,
                                                            (short)0, cin.v, false, false);
      }
      // ReLU + packed bf16 cvt, re-layout C(lane=N) -> A(lane=M) via per-wave LDS staging
#pragma unroll
      for (int n = 0; n < 4; ++n)
#pragma unroll
        for (int r = 0; r < 8; r += 2)
          relu_pair_bf16(&s.stage[wave][(r + hl * 8) * 64 + n * 16 + ln],
                         &s.stage[wave][(r + 1 + hl * 8) * 64 + n * 16 + ln],
                         accA[n].f[r], accA[n].f[r + 1]);
      asm volatile("s_wait_dscnt 0" ::: "memory");   // intra-wave LDS RAW

      // ---- layer 2: [16x64] @ [64x64] ----
      CF acc2[4];
#pragma unroll
      for (int n = 0; n < 4; ++n)
#pragma unroll
        for (int r = 0; r < 8; ++r) acc2[n].f[r] = bs2[mk][n];
#pragma unroll
      for (int kf = 0; kf < 2; ++kf) {
        AF a2;
        a2.h[0] = *(const v8bf*)&s.stage[wave][ln * 64 + kf * 32 + hl * 8];
        a2.h[1] = *(const v8bf*)&s.stage[wave][ln * 64 + kf * 32 + 16 + hl * 8];
#pragma unroll
        for (int n = 0; n < 4; ++n) {
          AF bfr;
          bfr.v = *(const v16bf*)&s.w.msg.w2sw[mk][kf][n][lane][0];  // contiguous 32B
          acc2[n].v = __builtin_amdgcn_wmma_f32_16x16x32_bf16(false, a2.v, false, bfr.v,
                                                              (short)0, acc2[n].v, false, false);
        }
      }
      // ReLU, weight by edge coefficient, segmented-reduce rows by receiver,
      // then at most 2 LDS atomics per n-tile (receiver-major edge order).
#pragma unroll
      for (int n = 0; n < 4; ++n) {
        float s0 = 0.0f, s1 = 0.0f;
#pragma unroll
        for (int r = 0; r < 8; ++r) {
          float sc = (mk == 0) ? c0v[r] : (1.0f - c0v[r]);
          float v = fmaxf(acc2[n].f[r], 0.0f) * sc;
          if (ebase + r < eB) s0 += v; else s1 += v;
        }
        atomicAdd(&s.agg[iA][n * 16 + ln], s0);
        if (mt * 16 + 15 >= eB) atomicAdd(&s.agg[iA + 1][n * 16 + ln], s1);
      }
    }
  }
  __syncthreads();

  // ---------- overlay: swizzle out-MLP weights into LDS (msg weights now dead) ----------
  for (int idx = tid; idx < 3 * 4 * 32 * 16; idx += 256) {     // out_W1 padded 80->96
    int f = idx & 15, lanei = (idx >> 4) & 31, n = (idx >> 9) & 3, kf = idx >> 11;
    int kk  = kf * 32 + (f >> 3) * 16 + (lanei >> 4) * 8 + (f & 7);
    int col = n * 16 + (lanei & 15);
    s.w.outw.o1sw[kf][n][lanei][f] = (kk < 80) ? f2bf(out_W1[kk * 64 + col]) : (unsigned short)0;
  }
  for (int idx = tid; idx < 2 * 4 * 32 * 16; idx += 256) {     // out_W2
    int f = idx & 15, lanei = (idx >> 4) & 31, n = (idx >> 9) & 3, kf = idx >> 11;
    int kk  = kf * 32 + (f >> 3) * 16 + (lanei >> 4) * 8 + (f & 7);
    int col = n * 16 + (lanei & 15);
    s.w.outw.o2sw[kf][n][lanei][f] = f2bf(out_W2[kk * 64 + col]);
  }
  for (int idx = tid; idx < 2 * 32 * 16; idx += 256) {         // out_W3 (64x16)
    int f = idx & 15, lanei = (idx >> 4) & 31, kf = idx >> 9;
    int kk  = kf * 32 + (f >> 3) * 16 + (lanei >> 4) * 8 + (f & 7);
    int col = lanei & 15;
    s.w.outw.o3sw[kf][lanei][f] = f2bf(out_W3[kk * 16 + col]);
  }
  __syncthreads();

  // ---------- node output MLP: 32 agents = 2 M-tiles handled by waves 0,1 ----------
  if (wave < 2) {
    // aug = [xs(16) | agg(64) | zero-pad(16)] : 32x96 bf16, reuse stage[4..6]
    unsigned short* aug = &s.stage[4][0];
    const int base = wave * 16;
    for (int idx = lane; idx < 16 * 96; idx += 32) {
      int rr = idx / 96, cc = idx - rr * 96;
      int row = base + rr;
      unsigned short v;
      if (cc < 16)      v = s.xsb[row][cc];
      else if (cc < 80) v = f2bf(s.agg[row][cc - 16]);
      else              v = 0;
      aug[row * 96 + cc] = v;
    }
    asm volatile("s_wait_dscnt 0" ::: "memory");

    // ---- out layer 1: [16x96(pad of 80)] @ [96x64] ----
    CF acc[4];
#pragma unroll
    for (int n = 0; n < 4; ++n) {
      float bias = s.ob1[n * 16 + ln];
#pragma unroll
      for (int r = 0; r < 8; ++r) acc[n].f[r] = bias;
    }
#pragma unroll
    for (int kf = 0; kf < 3; ++kf) {
      AF a2;
      a2.h[0] = *(const v8bf*)&aug[(base + ln) * 96 + kf * 32 + hl * 8];
      a2.h[1] = *(const v8bf*)&aug[(base + ln) * 96 + kf * 32 + 16 + hl * 8];
#pragma unroll
      for (int n = 0; n < 4; ++n) {
        AF bfr;
        bfr.v = *(const v16bf*)&s.w.outw.o1sw[kf][n][lane][0];
        acc[n].v = __builtin_amdgcn_wmma_f32_16x16x32_bf16(false, a2.v, false, bfr.v,
                                                           (short)0, acc[n].v, false, false);
      }
    }
#pragma unroll
    for (int n = 0; n < 4; ++n)
#pragma unroll
      for (int r = 0; r < 8; r += 2)
        relu_pair_bf16(&s.stage[wave][(r + hl * 8) * 64 + n * 16 + ln],
                       &s.stage[wave][(r + 1 + hl * 8) * 64 + n * 16 + ln],
                       acc[n].f[r], acc[n].f[r + 1]);
    asm volatile("s_wait_dscnt 0" ::: "memory");

    // ---- out layer 2: [16x64] @ [64x64] ----
    CF acc2[4];
#pragma unroll
    for (int n = 0; n < 4; ++n) {
      float bias = s.ob2[n * 16 + ln];
#pragma unroll
      for (int r = 0; r < 8; ++r) acc2[n].f[r] = bias;
    }
#pragma unroll
    for (int kf = 0; kf < 2; ++kf) {
      AF a2;
      a2.h[0] = *(const v8bf*)&s.stage[wave][ln * 64 + kf * 32 + hl * 8];
      a2.h[1] = *(const v8bf*)&s.stage[wave][ln * 64 + kf * 32 + 16 + hl * 8];
#pragma unroll
      for (int n = 0; n < 4; ++n) {
        AF bfr;
        bfr.v = *(const v16bf*)&s.w.outw.o2sw[kf][n][lane][0];
        acc2[n].v = __builtin_amdgcn_wmma_f32_16x16x32_bf16(false, a2.v, false, bfr.v,
                                                            (short)0, acc2[n].v, false, false);
      }
    }
#pragma unroll
    for (int n = 0; n < 4; ++n)
#pragma unroll
      for (int r = 0; r < 8; r += 2)
        relu_pair_bf16(&s.stage[wave][(r + hl * 8) * 64 + n * 16 + ln],
                       &s.stage[wave][(r + 1 + hl * 8) * 64 + n * 16 + ln],
                       acc2[n].f[r], acc2[n].f[r + 1]);
    asm volatile("s_wait_dscnt 0" ::: "memory");

    // ---- out layer 3: [16x64] @ [64x16], residual, store ----
    CF acc3;
#pragma unroll
    for (int r = 0; r < 8; ++r) acc3.f[r] = s.ob3[ln];
#pragma unroll
    for (int kf = 0; kf < 2; ++kf) {
      AF a2;
      a2.h[0] = *(const v8bf*)&s.stage[wave][ln * 64 + kf * 32 + hl * 8];
      a2.h[1] = *(const v8bf*)&s.stage[wave][ln * 64 + kf * 32 + 16 + hl * 8];
      AF bfr;
      bfr.v = *(const v16bf*)&s.w.outw.o3sw[kf][lane][0];
      acc3.v = __builtin_amdgcn_wmma_f32_16x16x32_bf16(false, a2.v, false, bfr.v,
                                                       (short)0, acc3.v, false, false);
    }
    const int tau = 2 * t + STEP;
    if (STEP == 0 || t < 31) {   // tau=63 is truncated by [:, :T-1]
#pragma unroll
      for (int r = 0; r < 8; ++r) {
        int aRow = base + r + hl * 8;
        float v = acc3.f[r] + s.xs[aRow][ln];
        outp[((b * An + aRow) * (Tn - 1) + tau) * Dn + ln] = v;
      }
    }
  }
}

} // namespace

extern "C" void kernel_launch(void* const* d_in, const int* in_sizes, int n_in,
                              void* d_out, int out_size, void* d_ws, size_t ws_size,
                              hipStream_t stream) {
  (void)in_sizes; (void)n_in; (void)d_ws; (void)ws_size; (void)out_size;
  const float* inputs    = (const float*)d_in[0];
  // d_in[1] rel_rec, d_in[2] rel_send: encoded analytically (receiver-major edge order)
  const float* rel_graph = (const float*)d_in[3];
  const float* gumbel    = (const float*)d_in[4];
  const float* msg_W1    = (const float*)d_in[5];
  const float* msg_b1    = (const float*)d_in[6];
  const float* msg_W2    = (const float*)d_in[7];
  const float* msg_b2    = (const float*)d_in[8];
  const float* out_W1    = (const float*)d_in[9];
  const float* out_b1    = (const float*)d_in[10];
  const float* out_W2    = (const float*)d_in[11];
  const float* out_b2    = (const float*)d_in[12];
  const float* out_W3    = (const float*)d_in[13];
  const float* out_b3    = (const float*)d_in[14];
  float* outp = (float*)d_out;

  dim3 grid(32 * 32), block(256);
  nri_step_kernel<0><<<grid, block, 0, stream>>>(inputs, outp, rel_graph, gumbel,
      msg_W1, msg_b1, msg_W2, msg_b2, out_W1, out_b1, out_W2, out_b2, out_W3, out_b3);
  nri_step_kernel<1><<<grid, block, 0, stream>>>(inputs, outp, rel_graph, gumbel,
      msg_W1, msg_b1, msg_W2, msg_b2, out_W1, out_b1, out_W2, out_b2, out_W3, out_b3);
}
